// TransformerLinearModel_RP_49804440765162
// MI455X (gfx1250) — matrix operand order repaired and checked by reading, not compiled, not run on hardware
//
#include <hip/hip_runtime.h>

// ---- problem constants -----------------------------------------------------
#define Bb     8
#define Tt     1024
#define IN_DIM 345
#define INP    352          // IN_DIM padded to a multiple of 32 (and 8)
#define Dd     512
#define Hh     8
#define Ll     6
#define DFFd   2048
#define NSPKd  4
#define DHd    64
#define GAPC   100

typedef __attribute__((ext_vector_type(16))) __bf16 v16bf;
typedef __attribute__((ext_vector_type(8)))  float  v8f;
typedef unsigned short u16;

// fragment-ready LDS tile: 32 lanes x (16 bf16 + 8 pad) = 768 ushorts (1.5KB)
#define FRAG_STRIDE 24
#define FRAG_TILE   768

// ---- bf16 helpers ----------------------------------------------------------
__device__ __forceinline__ u16 f2bfbits(float f) {
  union { float f; unsigned u; } a; a.f = f;
  unsigned r = a.u + 0x7FFFu + ((a.u >> 16) & 1u);   // round-to-nearest-even
  return (u16)(r >> 16);
}
__device__ __forceinline__ float bf2f(u16 h) {
  union { unsigned u; float f; } a; a.u = ((unsigned)h) << 16;
  return a.f;
}
__device__ __forceinline__ unsigned pack2bf(float a, float b) {
  return (unsigned)f2bfbits(a) | ((unsigned)f2bfbits(b) << 16);
}

__device__ __forceinline__ v8f wmma_bf16(v16bf a, v16bf b, v8f c) {
  return __builtin_amdgcn_wmma_f32_16x16x32_bf16(false, a, false, b, (short)0,
                                                 c, false, false);
}

// ---- fragment loads: lane's 16 operand bf16 are contiguous ------------------
__device__ __forceinline__ v16bf frag_ld(const u16* base) {     // LDS, padded
  const uint4* p = (const uint4*)(base + (threadIdx.x & 31) * FRAG_STRIDE);
  union { v16bf v; uint4 q[2]; } u;
  u.q[0] = p[0];
  u.q[1] = p[1];
  return u.v;
}
__device__ __forceinline__ v16bf frag_ld_g(const u16* base) {   // global, dense
  const uint4* p = (const uint4*)(base + (threadIdx.x & 31) * 16);
  union { v16bf v; uint4 q[2]; } u;
  u.q[0] = p[0];
  u.q[1] = p[1];
  return u.v;
}

// swizzle positions (this file's contiguous k order, k = 16*half + elem); for k % 4 == 0 a quad
// of consecutive k stays in one lane at consecutive element slots (8B store).
// A 16x32: lane = (r&15) + 16*(k>>4) ; elem = k&15
__device__ __forceinline__ int a_sw(int r, int k) {
  return ((r & 15) + ((k >> 4) << 4)) * FRAG_STRIDE +
         (k & 15);
}
// B 32x16: lane = (n&15) + 16*(k>>4) ; elem = k&15
__device__ __forceinline__ int b_sw(int k, int n) {
  return ((n & 15) + ((k >> 4) << 4)) * FRAG_STRIDE + (k & 15);
}

// ---- f32 -> bf16 conversion, row-major with zero padding (for A tensors) ---
__global__ __launch_bounds__(256) void conv2d_bf16_kernel(
    const float* __restrict__ in, u16* __restrict__ out,
    int in_rows, int in_cols, int out_rows, int out_cols) {
  long i = (long)blockIdx.x * 256 + threadIdx.x;        // pair index
  long total = (long)out_rows * out_cols / 2;
  if (i >= total) return;
  int c = (int)((i * 2) % out_cols), r = (int)((i * 2) / out_cols);
  float v0 = (r < in_rows && c < in_cols) ? in[(long)r * in_cols + c] : 0.f;
  float v1 = (r < in_rows && c + 1 < in_cols) ? in[(long)r * in_cols + c + 1]
                                              : 0.f;
  *(unsigned*)(out + (long)r * out_cols + c) = pack2bf(v0, v1);
}

// ---- f32 -> bf16 weight packer: emits WMMA B-fragment layout ---------------
// out[(k>>5)*(N/16) + (n>>4)][lane=(n&15)+16*((k&31)>>4)][elem=k&15]
__global__ __launch_bounds__(256) void conv_pack_w_kernel(
    const float* __restrict__ in, u16* __restrict__ out,
    int K, int N, int Kpad) {
  long i = (long)blockIdx.x * 256 + threadIdx.x;        // (k-pair, n) index
  long total = (long)(Kpad >> 1) * N;
  if (i >= total) return;
  int n = (int)(i % N);
  int k = (int)(i / N) * 2;
  float v0 = (k < K) ? in[(long)k * N + n] : 0.f;
  float v1 = (k + 1 < K) ? in[(long)(k + 1) * N + n] : 0.f;
  int kk = k & 31;
  long pos = ((long)(k >> 5) * (N >> 4) + (n >> 4)) * 512 +
             (((n & 15) + ((kk >> 4) << 4)) << 4) + (kk & 15);
  *(unsigned*)(out + pos) = pack2bf(v0, v1);
}

// ---- GEMM: C[M,N] = A[M,K] @ Wp + bias. A via LDS; B direct from packed ----
// global (L2-resident weights). block 128x128x32, 8 waves (4x2), wave 32x64.
#define BM  128
#define BN  128
#define BKK 32
__global__ __launch_bounds__(256) void gemm_bf16_kernel(
    const u16* __restrict__ A, int lda, const u16* __restrict__ Wp, int wtiles,
    const float* __restrict__ bias, float* __restrict__ Cf,
    u16* __restrict__ Cb, int ldc, int K, int relu) {
  __shared__ u16 As_sw[2][8 * FRAG_TILE];
  int bm = blockIdx.x * BM, bn = blockIdx.y * BN;
  int tid = threadIdx.x, lane = tid & 31, wave = tid >> 5;
  int wr = wave >> 1, wc = wave & 1;                // wave origin (wr*32, wc*64)
  v8f acc[2][4] = {};
  int nchunks = (K + BKK - 1) / BKK;                // operands are zero-padded

  int ra = tid >> 3, kqa = (tid & 7) << 2;          // A: row 0..31, k-quad

  auto stageA = [&](int ci, int p) {
    const u16* ag = A + (long)(bm + ra) * lda + ci * BKK + kqa;
    u16* as = &As_sw[p][(ra >> 4) * FRAG_TILE + a_sw(ra & 15, kqa)];
#pragma unroll
    for (int it = 0; it < 4; ++it)                  // rows ra + 32*it
      *(uint2*)(as + it * 2 * FRAG_TILE) =
          *(const uint2*)(ag + (long)it * 32 * lda);
  };

  stageA(0, 0);
  __syncthreads();
  for (int ci = 0; ci < nchunks; ++ci) {
    int p = ci & 1;
    if (ci + 1 < nchunks) stageA(ci + 1, p ^ 1);    // overlap with WMMA below
    if (ci + 2 < nchunks)                            // global_prefetch_b8
      __builtin_prefetch(&A[(long)(bm + (tid >> 1)) * lda + (ci + 2) * BKK],
                         0, 1);
    // B fragments straight from packed global weights (L2 resident)
    const u16* wbase = Wp + ((long)ci * wtiles + (bn >> 4) + wc * 4) * 512;
    v16bf af[2], bf[4];
#pragma unroll
    for (int tj = 0; tj < 4; ++tj)
      bf[tj] = frag_ld_g(wbase + tj * 512);
#pragma unroll
    for (int ti = 0; ti < 2; ++ti)
      af[ti] = frag_ld(As_sw[p] + (wr * 2 + ti) * FRAG_TILE);
#pragma unroll
    for (int ti = 0; ti < 2; ++ti)
#pragma unroll
      for (int tj = 0; tj < 4; ++tj)
        acc[ti][tj] = wmma_bf16(af[ti], bf[tj], acc[ti][tj]);
    __syncthreads();
  }

  // C/D f32 layout: lane<16 -> N=lane, M=i ; lane>=16 -> N=lane-16, M=8+i
  int nloc = lane & 15, mbase = (lane >> 4) << 3;
#pragma unroll
  for (int ti = 0; ti < 2; ++ti)
#pragma unroll
    for (int tj = 0; tj < 4; ++tj)
#pragma unroll
      for (int i = 0; i < 8; ++i) {
        int gr = bm + wr * 32 + ti * 16 + mbase + i;
        int gc = bn + wc * 64 + tj * 16 + nloc;
        float val = acc[ti][tj][i] + bias[gc];
        if (relu) val = fmaxf(val, 0.f);
        if (Cf) Cf[(long)gr * ldc + gc] = val;
        if (Cb) Cb[(long)gr * ldc + gc] = f2bfbits(val);
      }
}

// ---- layernorm over D=512, optional residual; writes f32 + bf16 shadow -----
__global__ __launch_bounds__(256) void ln_kernel(
    const float* __restrict__ x, const float* __restrict__ res,
    const float* __restrict__ g, const float* __restrict__ bta,
    float* __restrict__ out, u16* __restrict__ outb) {
  int row = blockIdx.x;
  int tid = threadIdx.x, lane = tid & 31, wave = tid >> 5;
  const float* xr = x + (long)row * Dd;
  float e0 = xr[tid], e1 = xr[tid + 256];
  if (res) {
    const float* rr = res + (long)row * Dd;
    e0 += rr[tid]; e1 += rr[tid + 256];
  }
  __shared__ float wsum[8];
  float s = e0 + e1;
#pragma unroll
  for (int o = 16; o; o >>= 1) s += __shfl_xor(s, o, 32);
  if (lane == 0) wsum[wave] = s;
  __syncthreads();
  float tot = 0;
#pragma unroll
  for (int i = 0; i < 8; ++i) tot += wsum[i];
  float mean = tot * (1.f / 512.f);
  float d0 = e0 - mean, d1 = e1 - mean;
  float sq = d0 * d0 + d1 * d1;
#pragma unroll
  for (int o = 16; o; o >>= 1) sq += __shfl_xor(sq, o, 32);
  __syncthreads();
  if (lane == 0) wsum[wave] = sq;
  __syncthreads();
  float v2 = 0;
#pragma unroll
  for (int i = 0; i < 8; ++i) v2 += wsum[i];
  float inv = rsqrtf(v2 * (1.f / 512.f) + 1e-5f);
  float o0 = d0 * inv * g[tid] + bta[tid];
  float o1 = d1 * inv * g[tid + 256] + bta[tid + 256];
  out[(long)row * Dd + tid] = o0;
  out[(long)row * Dd + tid + 256] = o1;
  outb[(long)row * Dd + tid] = f2bfbits(o0);
  outb[(long)row * Dd + tid + 256] = f2bfbits(o1);
}

// ---- fused attention: one block per (b, h, 32-query tile); bf16 q/k/v ------
__device__ __forceinline__ int bucket5(int d) {
  int q = (d >= 0) ? (d / GAPC) : -((-d + GAPC - 1) / GAPC);
  q = max(-2, min(2, q));
  return q + 2;
}

// dynamic LDS layout (bytes). CDNA5 WGP = 320KB, this uses ~251KB.
#define OFF_S    0                       // 32*1024 f32 scores     = 131072
#define OFF_PSW  131072                  // 64 A-frag prob tiles   =  98304
#define OFF_QSW  229376                  // 4 A-frag tiles (Q)     =   6144
#define OFF_TSW  235520                  // 8 B-frag tiles (K^T/V) =  12288
#define OFF_QR   247808                  // 32*8  f32              =   1024
#define OFF_PMAX 248832                  // 256   f32              =   1024
#define OFF_PSUM 249856                  // 256   f32              =   1024
#define OFF_PB   250880                  // 32*8*5 f32             =   5120
#define OFF_RSUM 256000                  // 32    f32              =    128
#define OFF_BSR  256128                  // 32*5  f32              =    640
#define ATTN_SMEM_BYTES 256768

__global__ __launch_bounds__(256) void attention_kernel(
    const u16* __restrict__ qm, const u16* __restrict__ km,
    const u16* __restrict__ vm, const float* __restrict__ relk,
    const float* __restrict__ relv, const int* __restrict__ slens,
    u16* __restrict__ omb) {
  extern __shared__ char smem[];
  float* Sl   = (float*)(smem + OFF_S);             // [32][1024] raw scores
  u16*   Psw  = (u16*)(smem + OFF_PSW);             // bf16 prob A-frag tiles
  u16*   Qsw  = (u16*)(smem + OFF_QSW);
  u16*   Tsw  = (u16*)(smem + OFF_TSW);
  float* qr   = (float*)(smem + OFF_QR);
  float* pmax = (float*)(smem + OFF_PMAX);
  float* psum = (float*)(smem + OFF_PSUM);
  float* pb   = (float*)(smem + OFF_PB);
  float* rsum = (float*)(smem + OFF_RSUM);
  float* bsr  = (float*)(smem + OFF_BSR);
  unsigned* Psw32 = (unsigned*)Psw;

  int b = blockIdx.z, h = blockIdx.y, q0 = blockIdx.x * 32;
  int tid = threadIdx.x, lane = tid & 31, wave = tid >> 5;
  int tr = wave >> 2, tc = wave & 3;                // wave tile in 32x64 grid
  int sl = slens[b];
  int nloc = lane & 15, mbase = (lane >> 4) << 3;

  // staging geometry (iteration-invariant per thread)
  int r16 = tid >> 4, dq = (tid & 15) << 2;         // Q/K: row|key, d-quad
  int dv = tid & 63, kq0 = (tid >> 6) << 2;         // V: d col, key-quad base

  // phase 0: Q tile -> A-frag tiles (straight-line 8B copies) + rel-k bias
  {
    const u16* qg = qm + ((long)(b * Tt + q0 + r16)) * Dd + h * DHd + dq;
    u16* qs = &Qsw[(dq >> 5) * FRAG_TILE + a_sw(r16 & 15, dq & 31)];
#pragma unroll
    for (int it = 0; it < 2; ++it)                  // rows r16 + 16*it
      *(uint2*)(qs + it * 2 * FRAG_TILE) =
          *(const uint2*)(qg + (long)it * 16 * Dd);
  }
  for (int job = tid; job < 32 * 5; job += 256) {
    int r = job / 5, m = job % 5;
    const u16* qrow = qm + ((long)(b * Tt + q0 + r)) * Dd + h * DHd;
    float acc = 0.f;
#pragma unroll 8
    for (int d = 0; d < DHd; ++d) acc += bf2f(qrow[d]) * relk[m * DHd + d];
    qr[r * 8 + m] = acc;
  }
  __syncthreads();

  // phase 1: full score strip S[32][1024] via WMMA over 16 key chunks of 64
  for (int kc = 0; kc < 16; ++kc) {
    {
      const u16* kg = km + ((long)(b * Tt + kc * 64 + r16)) * Dd + h * DHd + dq;
      u16* ks = &Tsw[(dq >> 5) * FRAG_TILE + b_sw(dq & 31, r16 & 15)];
#pragma unroll
      for (int it = 0; it < 4; ++it)                // keys r16 + 16*it
        *(uint2*)(ks + it * 2 * FRAG_TILE) =
            *(const uint2*)(kg + (long)it * 16 * Dd);
    }
    __syncthreads();
    v8f s = {};
#pragma unroll
    for (int ks = 0; ks < 2; ++ks) {
      v16bf aq = frag_ld(Qsw + (tr * 2 + ks) * FRAG_TILE);
      v16bf bk = frag_ld(Tsw + (tc * 2 + ks) * FRAG_TILE);
      s = wmma_bf16(aq, bk, s);
    }
#pragma unroll
    for (int i = 0; i < 8; ++i) {
      int row  = tr * 16 + mbase + i;
      int colg = kc * 64 + tc * 16 + nloc;
      int qg   = q0 + row;
      float val = (s[i] + qr[row * 8 + bucket5(colg - qg)]) * 0.125f;
      if (colg >= sl) val = -1e9f;
      Sl[row * 1024 + colg] = val;
    }
    __syncthreads();
  }

  // phase 2: softmax; write probs as bf16 directly in A-fragment swizzle
  {
    int row = tid >> 3, part = tid & 7, c0 = part * 128;
    int qg = q0 + row;
    float mx = -3.4e38f;
    for (int c = c0; c < c0 + 128; ++c) mx = fmaxf(mx, Sl[row * 1024 + c]);
    pmax[tid] = mx;
    __syncthreads();
    float rmx = -3.4e38f;
#pragma unroll
    for (int p = 0; p < 8; ++p) rmx = fmaxf(rmx, pmax[row * 8 + p]);
    float sum = 0.f, bs[5] = {0.f, 0.f, 0.f, 0.f, 0.f};
    int rowtile = (row >> 4) << 5;                  // 32 tiles per row-tile
    for (int c = c0; c < c0 + 128; c += 2) {
      float e0 = __expf(Sl[row * 1024 + c] - rmx);
      float e1 = __expf(Sl[row * 1024 + c + 1] - rmx);
      sum += e0 + e1;
      bs[bucket5(c - qg)] += e0;
      bs[bucket5(c + 1 - qg)] += e1;
      int tile = rowtile + ((c >> 6) << 1) + ((c >> 5) & 1);
      Psw32[(tile * FRAG_TILE + a_sw(row & 15, c & 31)) >> 1] = pack2bf(e0, e1);
    }
    psum[tid] = sum;
#pragma unroll
    for (int m = 0; m < 5; ++m) pb[tid * 5 + m] = bs[m];
    __syncthreads();
    if (part == 0) {
      float t = 0.f;
#pragma unroll
      for (int p = 0; p < 8; ++p) t += psum[row * 8 + p];
      rsum[row] = t;
#pragma unroll
      for (int m = 0; m < 5; ++m) {
        float u = 0.f;
#pragma unroll
        for (int p = 0; p < 8; ++p) u += pb[(row * 8 + p) * 5 + m];
        bsr[row * 5 + m] = u;
      }
    }
    __syncthreads();
  }

  // phase 3: O = P @ V, both operands fragment-ready in LDS
  v8f oacc = {};
  for (int kc = 0; kc < 16; ++kc) {
    {
      const u16* vg = vm + ((long)(b * Tt + kc * 64 + kq0)) * Dd + h * DHd + dv;
      u16* vs = &Tsw[((dv >> 4) << 1) * FRAG_TILE + (dv & 15) * FRAG_STRIDE +
                     kq0];
#pragma unroll
      for (int it = 0; it < 4; ++it) {              // keys kq0 + 16*it
        const u16* vp = vg + (long)(16 * it) * Dd;
        uint2 pk;
        pk.x = (unsigned)vp[0] | ((unsigned)vp[Dd] << 16);
        pk.y = (unsigned)vp[2 * Dd] | ((unsigned)vp[3 * Dd] << 16);
        *(uint2*)(vs + (it >> 1) * FRAG_TILE + (it & 1) * (16 * FRAG_STRIDE)) =
            pk;
      }
    }
    __syncthreads();
#pragma unroll
    for (int ks = 0; ks < 2; ++ks) {
      v16bf ap = frag_ld(Psw + ((tr << 5) + (kc << 1) + ks) * FRAG_TILE);
      v16bf bv = frag_ld(Tsw + (tc * 2 + ks) * FRAG_TILE);
      oacc = wmma_bf16(ap, bv, oacc);
    }
    __syncthreads();
  }

  // epilogue: + bucket_sums * rel_v, normalize, store bf16 (O-proj operand)
#pragma unroll
  for (int i = 0; i < 8; ++i) {
    int row  = tr * 16 + mbase + i;
    int dcol = tc * 16 + nloc;
    float val = oacc[i];
#pragma unroll
    for (int m = 0; m < 5; ++m) val += bsr[row * 5 + m] * relv[m * DHd + dcol];
    val /= rsum[row];
    omb[((long)(b * Tt + q0 + row)) * Dd + h * DHd + dcol] = f2bfbits(val);
  }
}

// ---- decoder: sigmoid(x @ dec_W + dec_b), N=4 ------------------------------
__global__ __launch_bounds__(256) void decoder_kernel(
    const float* __restrict__ x, const float* __restrict__ W,
    const float* __restrict__ bias, float* __restrict__ out) {
  int i = blockIdx.x * 256 + threadIdx.x;
  if (i >= Bb * Tt * NSPKd) return;
  int row = i >> 2, s = i & 3;
  float acc = bias[s];
  const float* xr = x + (long)row * Dd;
#pragma unroll 8
  for (int d = 0; d < Dd; ++d) acc += xr[d] * W[d * NSPKd + s];
  out[i] = 1.f / (1.f + __expf(-acc));
}

// ---- host launcher ---------------------------------------------------------
extern "C" void kernel_launch(void* const* d_in, const int* in_sizes, int n_in,
                              void* d_out, int out_size, void* d_ws,
                              size_t ws_size, hipStream_t stream) {
  (void)in_sizes; (void)n_in; (void)out_size; (void)ws_size;
  const float* src    = (const float*)d_in[0];
  const float* enc_W  = (const float*)d_in[1];
  const float* enc_b  = (const float*)d_in[2];
  const float* encn_g = (const float*)d_in[3];
  const float* encn_b = (const float*)d_in[4];
  const float* Wq = (const float*)d_in[5];
  const float* bq = (const float*)d_in[6];
  const float* Wk = (const float*)d_in[7];
  const float* bk = (const float*)d_in[8];
  const float* Wv = (const float*)d_in[9];
  const float* bv = (const float*)d_in[10];
  const float* Wo = (const float*)d_in[11];
  const float* bo = (const float*)d_in[12];
  const float* rel_k = (const float*)d_in[13];
  const float* rel_v = (const float*)d_in[14];
  const float* W1 = (const float*)d_in[15];
  const float* b1 = (const float*)d_in[16];
  const float* W2 = (const float*)d_in[17];
  const float* b2 = (const float*)d_in[18];
  const float* n1g = (const float*)d_in[19];
  const float* n1b = (const float*)d_in[20];
  const float* n2g = (const float*)d_in[21];
  const float* n2b = (const float*)d_in[22];
  const float* dec_W = (const float*)d_in[23];
  const float* dec_b = (const float*)d_in[24];
  const int* seq_lens = (const int*)d_in[25];
  float* out = (float*)d_out;

  const long RT = (long)Bb * Tt;                    // 8192 rows
  char* w = (char*)d_ws;
  float* x  = (float*)w;            w += RT * Dd * 4;       // f32 activations
  float* rb = (float*)w;            w += RT * Dd * 4;       // f32 residual
  u16* xb   = (u16*)w;              w += RT * Dd * 2;       // bf16 shadows:
  u16* qbb  = (u16*)w;              w += RT * Dd * 2;
  u16* kbb  = (u16*)w;              w += RT * Dd * 2;
  u16* vbb  = (u16*)w;              w += RT * Dd * 2;
  u16* obb  = (u16*)w;              w += RT * Dd * 2;
  u16* ffb  = (u16*)w;              w += RT * DFFd * 2;
  u16* srcb = (u16*)w;              w += RT * INP * 2;      // src padded ->352
  u16* eWb  = (u16*)w;              w += (long)INP * Dd * 2;  // packed weights
  u16* Wqb  = (u16*)w;              w += (long)Ll * Dd * Dd * 2;
  u16* Wkb  = (u16*)w;              w += (long)Ll * Dd * Dd * 2;
  u16* Wvb  = (u16*)w;              w += (long)Ll * Dd * Dd * 2;
  u16* Wob  = (u16*)w;              w += (long)Ll * Dd * Dd * 2;
  u16* W1b  = (u16*)w;              w += (long)Ll * Dd * DFFd * 2;
  u16* W2b  = (u16*)w;              w += (long)Ll * DFFd * Dd * 2;

  hipFuncSetAttribute((const void*)attention_kernel,
                      hipFuncAttributeMaxDynamicSharedMemorySize,
                      ATTN_SMEM_BYTES);

  dim3 blk(256);
  auto cg = [](long total) { return dim3((unsigned)((total + 255) / 256)); };

  // producer-side conversions: src row-major padded; weights frag-packed
  conv2d_bf16_kernel<<<cg(RT * INP / 2), blk, 0, stream>>>(
      src, srcb, (int)RT, IN_DIM, (int)RT, INP);
  conv_pack_w_kernel<<<cg((long)(INP / 2) * Dd), blk, 0, stream>>>(
      enc_W, eWb, IN_DIM, Dd, INP);
  for (int l = 0; l < Ll; ++l) {
    long oqk = (long)l * Dd * Dd, off1 = (long)l * Dd * DFFd;
    conv_pack_w_kernel<<<cg((long)(Dd / 2) * Dd), blk, 0, stream>>>(
        Wq + oqk, Wqb + oqk, Dd, Dd, Dd);
    conv_pack_w_kernel<<<cg((long)(Dd / 2) * Dd), blk, 0, stream>>>(
        Wk + oqk, Wkb + oqk, Dd, Dd, Dd);
    conv_pack_w_kernel<<<cg((long)(Dd / 2) * Dd), blk, 0, stream>>>(
        Wv + oqk, Wvb + oqk, Dd, Dd, Dd);
    conv_pack_w_kernel<<<cg((long)(Dd / 2) * Dd), blk, 0, stream>>>(
        Wo + oqk, Wob + oqk, Dd, Dd, Dd);
    conv_pack_w_kernel<<<cg((long)(Dd / 2) * DFFd), blk, 0, stream>>>(
        W1 + off1, W1b + off1, Dd, DFFd, Dd);
    conv_pack_w_kernel<<<cg((long)(DFFd / 2) * Dd), blk, 0, stream>>>(
        W2 + off1, W2b + off1, DFFd, Dd, DFFd);
  }

  dim3 gD(RT / BM, Dd / BN);                 // (64, 4)
  dim3 gF(RT / BM, DFFd / BN);               // (64, 16)

  // encoder + layernorm
  gemm_bf16_kernel<<<gD, blk, 0, stream>>>(srcb, INP, eWb, Dd / 16, enc_b,
                                           x, nullptr, Dd, IN_DIM, 0);
  ln_kernel<<<RT, blk, 0, stream>>>(x, nullptr, encn_g, encn_b, x, xb);

  for (int l = 0; l < Ll; ++l) {
    long oqk = (long)l * Dd * Dd, off1 = (long)l * Dd * DFFd;
    gemm_bf16_kernel<<<gD, blk, 0, stream>>>(xb, Dd, Wqb + oqk, Dd / 16,
                                             bq + l * Dd, nullptr, qbb, Dd,
                                             Dd, 0);
    gemm_bf16_kernel<<<gD, blk, 0, stream>>>(xb, Dd, Wkb + oqk, Dd / 16,
                                             bk + l * Dd, nullptr, kbb, Dd,
                                             Dd, 0);
    gemm_bf16_kernel<<<gD, blk, 0, stream>>>(xb, Dd, Wvb + oqk, Dd / 16,
                                             bv + l * Dd, nullptr, vbb, Dd,
                                             Dd, 0);

    attention_kernel<<<dim3(Tt / 32, Hh, Bb), blk, ATTN_SMEM_BYTES, stream>>>(
        qbb, kbb, vbb, rel_k + (long)l * 5 * DHd, rel_v + (long)l * 5 * DHd,
        seq_lens, obb);

    gemm_bf16_kernel<<<gD, blk, 0, stream>>>(obb, Dd, Wob + oqk, Dd / 16,
                                             bo + l * Dd, rb, nullptr, Dd,
                                             Dd, 0);
    ln_kernel<<<RT, blk, 0, stream>>>(x, rb, n1g + l * Dd, n1b + l * Dd, x, xb);

    gemm_bf16_kernel<<<gF, blk, 0, stream>>>(xb, Dd, W1b + off1, DFFd / 16,
                                             b1 + l * DFFd, nullptr, ffb, DFFd,
                                             Dd, 1);
    gemm_bf16_kernel<<<gD, blk, 0, stream>>>(ffb, DFFd, W2b + off1, Dd / 16,
                                             b2 + l * Dd, rb, nullptr, Dd,
                                             DFFd, 0);
    ln_kernel<<<RT, blk, 0, stream>>>(x, rb, n2g + l * Dd, n2b + l * Dd, x, xb);
  }

  decoder_kernel<<<(Bb * Tt * NSPKd + 255) / 256, blk, 0, stream>>>(
      x, dec_W, dec_b, out);
}
